// DecoderLayer_23158463660754
// MI455X (gfx1250) — matrix operand order repaired and checked by reading, not compiled, not run on hardware
//
#include <hip/hip_runtime.h>
#include <hip/hip_bf16.h>

// ---------------------------------------------------------------------------
// Decoder layer for MI455X (gfx1250).
//  - All GEMM-like math: v_wmma_f32_16x16x32_bf16, fp32 accumulation.
//  - Global->LDS staging of bf16 tiles via global_load_async_to_lds_b128
//    (ASYNCcnt) with double buffering: copy of tile k+1 overlaps WMMA of k.
//  - Chunked causal linear attention with LDS-resident running state and
//    global_prefetch of the next chunk.
// ---------------------------------------------------------------------------

typedef __bf16 bf16_t;
typedef __attribute__((ext_vector_type(4)))  bf16_t v4bf;
typedef __attribute__((ext_vector_type(8)))  bf16_t v8bf;
typedef __attribute__((ext_vector_type(16))) bf16_t v16bf;
typedef __attribute__((ext_vector_type(8)))  float  v8f;

static __device__ __forceinline__ bf16_t tobf(float f) { return (bf16_t)f; }

static __device__ __forceinline__ v8f v8f_zero() {
  v8f z = {0.f, 0.f, 0.f, 0.f, 0.f, 0.f, 0.f, 0.f};
  return z;
}

static __device__ __forceinline__ v16bf frag_cat(v8bf a, v8bf b) {
  return __builtin_shufflevector(a, b, 0, 1, 2, 3, 4, 5, 6, 7,
                                       8, 9, 10, 11, 12, 13, 14, 15);
}

// A fragment (16x32 bf16) from row-major [row][k] storage, 'stride' halves/row.
// CDNA5 layout: lanes 0-15 -> M=0..15 K={0..7,16..23}; lanes 16-31 -> K+8.
static __device__ __forceinline__ v16bf load_frag_a(const bf16_t* base, int stride,
                                                    int row0, int k0, int lane) {
  const int r  = row0 + (lane & 15);
  const int lo = (lane >> 4) << 3;  // 0 or 8
  const bf16_t* p = base + r * stride + k0 + lo;
  v8bf x = *(const v8bf*)(p);        // K = k0+lo .. +7
  v8bf y = *(const v8bf*)(p + 16);   // K = k0+16+lo .. +7
  return frag_cat(x, y);
}

// B fragment (32x16 bf16) from row-major [k][n] storage: lane -> N = col0+(lane&15),
// K = {0..7,16..23} for lanes 0-15, K+8 for lanes 16-31 (matches A).
static __device__ __forceinline__ v16bf load_frag_b(const bf16_t* base, int stride,
                                                    int k0, int col0, int lane) {
  const bf16_t* p = base + (k0 + ((lane >> 4) << 3)) * stride + col0 + (lane & 15);
  v16bf v;
#pragma unroll
  for (int s = 0; s < 8; ++s) {
    v[s]     = p[s * stride];
    v[8 + s] = p[(16 + s) * stride];
  }
  return v;
}

static __device__ __forceinline__ v8f wmma_bf16(v16bf a, v16bf b, v8f c) {
  return __builtin_amdgcn_wmma_f32_16x16x32_bf16(false, a, false, b,
                                                 (short)0, c, false, false);
}

// Issue one 16-byte async global->LDS copy (raw bytes, ASYNCcnt-tracked).
static __device__ __forceinline__ void async_b128(unsigned lds_addr,
                                                  unsigned gl_off,
                                                  unsigned long long gl_base) {
  asm volatile("global_load_async_to_lds_b128 %0, %1, %2"
               :: "v"(lds_addr), "v"(gl_off), "s"(gl_base) : "memory");
}
static __device__ __forceinline__ void wait_async_le4() {
  asm volatile("s_wait_asynccnt 0x4" ::: "memory");
}
static __device__ __forceinline__ void wait_async_0() {
  asm volatile("s_wait_asynccnt 0x0" ::: "memory");
}

// ---------------------------------------------------------------------------
// f32 -> bf16 bulk convert (4 elements / thread).
// ---------------------------------------------------------------------------
__global__ __launch_bounds__(256)
void cvt_f32_bf16_k(const float* __restrict__ in, bf16_t* __restrict__ out) {
  const size_t i = ((size_t)blockIdx.x * 256 + threadIdx.x) * 4;
  const float4 v = *(const float4*)(in + i);
  v4bf o = {tobf(v.x), tobf(v.y), tobf(v.z), tobf(v.w)};
  *(v4bf*)(out + i) = o;
}

// ---------------------------------------------------------------------------
// GEMM: C[M,N] = act(A[M,K] @ W[K,N] + bias[N]); bf16 in, fp32 accum.
// 256 threads (8 waves), 128x128 tile, BK=32, double-buffered async staging.
// Optional fp32 and/or bf16 outputs.
// ---------------------------------------------------------------------------
#define GA_STRIDE 40   // halves per A-tile row (32 + pad; 80B, 16B-aligned)
#define GB_STRIDE 136  // halves per B-tile row (128 + pad; 272B, 16B-aligned)

__global__ __launch_bounds__(256)
void gemm_bias_act(const bf16_t* __restrict__ A, const bf16_t* __restrict__ W,
                   const float* __restrict__ bias, float* __restrict__ C,
                   bf16_t* __restrict__ Cbf, int K, int N, int act) {
  __shared__ __attribute__((aligned(16))) bf16_t As[2][128 * GA_STRIDE];
  __shared__ __attribute__((aligned(16))) bf16_t Bs[2][32 * GB_STRIDE];
  const int tid = threadIdx.x, lane = tid & 31, wid = tid >> 5;
  const int wm = wid >> 2, wn = wid & 3;
  const int row0 = blockIdx.y * 128, col0 = blockIdx.x * 128;
  const unsigned long long gA = (unsigned long long)A;
  const unsigned long long gW = (unsigned long long)W;

  v8f acc[4][2];
#pragma unroll
  for (int i = 0; i < 4; ++i)
#pragma unroll
    for (int j = 0; j < 2; ++j) acc[i][j] = v8f_zero();

  // stage one K-tile (A:128x32, B:32x128 bf16) into buffer `buf`; 4 ops/thread
  auto stage = [&](int buf, int k0) {
#pragma unroll
    for (int i = 0; i < 2; ++i) {
      const int cid = i * 256 + tid;
      {  // A tile: rows of 64B = 4 chunks
        const int r = cid >> 2, q = cid & 3;
        const unsigned lds = (unsigned)(size_t)(&As[buf][r * GA_STRIDE]) + q * 16;
        const unsigned gof = (unsigned)(((row0 + r) * K + k0) * 2 + q * 16);
        async_b128(lds, gof, gA);
      }
      {  // B tile: rows of 256B = 16 chunks
        const int r = cid >> 4, q = cid & 15;
        const unsigned lds = (unsigned)(size_t)(&Bs[buf][r * GB_STRIDE]) + q * 16;
        const unsigned gof = (unsigned)(((k0 + r) * N + col0) * 2 + q * 16);
        async_b128(lds, gof, gW);
      }
    }
  };

  const int nk = K / 32;
  stage(0, 0);
  for (int kt = 0; kt < nk; ++kt) {
    const int cur = kt & 1;
    if (kt + 1 < nk) {
      stage(cur ^ 1, (kt + 1) * 32);
      wait_async_le4();   // in-order retire: current tile (4 ops) resident
    } else {
      wait_async_0();
    }
    __syncthreads();
    const bf16_t* Ab = As[cur];
    const bf16_t* Bb = Bs[cur];
    v16bf af[4], bfr[2];
#pragma unroll
    for (int tm = 0; tm < 4; ++tm)
      af[tm] = load_frag_a(Ab, GA_STRIDE, wm * 64 + tm * 16, 0, lane);
#pragma unroll
    for (int tn = 0; tn < 2; ++tn)
      bfr[tn] = load_frag_b(Bb, GB_STRIDE, 0, wn * 32 + tn * 16, lane);
#pragma unroll
    for (int tm = 0; tm < 4; ++tm)
#pragma unroll
      for (int tn = 0; tn < 2; ++tn)
        acc[tm][tn] = wmma_bf16(af[tm], bfr[tn], acc[tm][tn]);
    __syncthreads();
  }

#pragma unroll
  for (int tm = 0; tm < 4; ++tm)
#pragma unroll
    for (int tn = 0; tn < 2; ++tn) {
      const int rbase = row0 + wm * 64 + tm * 16 + ((lane >> 4) << 3);
      const int col   = col0 + wn * 32 + tn * 16 + (lane & 15);
      const float bv = bias[col];
#pragma unroll
      for (int r = 0; r < 8; ++r) {
        float v = acc[tm][tn][r] + bv;
        if (act) v = fmaxf(v, 0.0f);
        if (C)   C[(size_t)(rbase + r) * N + col] = v;
        if (Cbf) Cbf[(size_t)(rbase + r) * N + col] = tobf(v);
      }
    }
}

// ---------------------------------------------------------------------------
// Causal linear attention (bucket_size=1), chunked (C=64) per (b,h).
// qvk row layout: [q(512) | v(512) | k(512)], row stride 1536. fp32 in/out.
// ---------------------------------------------------------------------------
#define SQ 72  // LDS row stride in halves

__global__ __launch_bounds__(128)
void causal_linear_attn_k(const float* __restrict__ qvk, float* __restrict__ out) {
  __shared__ __attribute__((aligned(16))) bf16_t qs_s[64 * SQ];
  __shared__ __attribute__((aligned(16))) bf16_t ksT_s[64 * SQ];   // [e][m]
  __shared__ __attribute__((aligned(16))) bf16_t v_s[64 * SQ];     // [m][e]
  __shared__ __attribute__((aligned(16))) bf16_t S_s[64 * SQ];     // masked scores
  __shared__ __attribute__((aligned(16))) bf16_t ctxb_s[64 * SQ];  // bf16 ctx [d][e]
  __shared__ float ctx_s[64 * 68];                                  // fp32 ctx [d][e]
  __shared__ float kcum_s[64];
  __shared__ float dinv_s[64];

  const int tid = threadIdx.x, lane = tid & 31, wid = tid >> 5;
  const int b = blockIdx.x >> 3, h = blockIdx.x & 7;

  for (int t = tid; t < 64 * 68; t += 128) ctx_s[t] = 0.f;
  for (int t = tid; t < 64 * SQ; t += 128) ctxb_s[t] = tobf(0.f);
  if (tid < 64) kcum_s[tid] = 0.f;
  __syncthreads();

  for (int n0 = 0; n0 < 2048; n0 += 64) {
    // ---- prefetch next chunk (hide HBM latency in this serial scan)
    if (n0 + 64 < 2048 && tid < 64) {
      const float* nx = qvk + (size_t)(b * 2048 + n0 + 64 + tid) * 1536 + h * 64;
      __builtin_prefetch(nx, 0, 1);
      __builtin_prefetch(nx + 32, 0, 1);
      __builtin_prefetch(nx + 512, 0, 1);
      __builtin_prefetch(nx + 544, 0, 1);
      __builtin_prefetch(nx + 1024, 0, 1);
      __builtin_prefetch(nx + 1056, 0, 1);
    }
    // ---- prep: ksT = exp(k)^T, v copy, qs = softmax(q) * E^-0.5
    for (int t = tid; t < 4096; t += 128) {
      const int m = t >> 6, e = t & 63;
      const size_t rb = (size_t)(b * 2048 + n0 + m) * 1536 + h * 64 + e;
      ksT_s[e * SQ + m] = tobf(__expf(qvk[rb + 1024]));
      v_s[m * SQ + e]   = tobf(qvk[rb + 512]);
    }
    if (tid < 64) {
      const int m = tid;
      const size_t rb = (size_t)(b * 2048 + n0 + m) * 1536 + h * 64;
      float mx = -3.4e38f;
      for (int e = 0; e < 64; ++e) mx = fmaxf(mx, qvk[rb + e]);
      float s = 0.f;
      for (int e = 0; e < 64; ++e) {
        const float ev = __expf(qvk[rb + e] - mx);
        s += ev;
        qs_s[m * SQ + e] = tobf(ev);
      }
      const float inv = 0.125f / s;  // E^-0.5 = 1/8
      for (int e = 0; e < 64; ++e)
        qs_s[m * SQ + e] = tobf((float)qs_s[m * SQ + e] * inv);
    }
    __syncthreads();

    // ---- S = tril(qs @ ks^T)   (16 tiles over 4 waves)
    for (int i = 0; i < 4; ++i) {
      const int idx = wid * 4 + i, tm = idx >> 2, tn = idx & 3;
      v8f c = v8f_zero();
      c = wmma_bf16(load_frag_a(qs_s, SQ, tm * 16, 0, lane),
                    load_frag_b(ksT_s, SQ, 0, tn * 16, lane), c);
      c = wmma_bf16(load_frag_a(qs_s, SQ, tm * 16, 32, lane),
                    load_frag_b(ksT_s, SQ, 32, tn * 16, lane), c);
#pragma unroll
      for (int r = 0; r < 8; ++r) {
        const int grow = tm * 16 + r + ((lane >> 4) << 3);
        const int gcol = tn * 16 + (lane & 15);
        S_s[grow * SQ + gcol] = tobf((gcol <= grow) ? c[r] : 0.f);
      }
    }
    __syncthreads();

    // ---- denominator: 1 / (rowsum(S) + qs . (kcum_prev + eps))
    if (tid < 64) {
      const int m = tid;
      float rs = 0.f;
      for (int j = 0; j < 64; ++j) rs += (float)S_s[m * SQ + j];
      float dt = 0.f;
      for (int e = 0; e < 64; ++e)
        dt += (float)qs_s[m * SQ + e] * (kcum_s[e] + 1e-6f);
      dinv_s[m] = 1.0f / (rs + dt);
    }
    __syncthreads();
    if (tid < 64) {  // kcum += colsum(ks)
      const int e = tid;
      float s = kcum_s[e];
      for (int m = 0; m < 64; ++m) s += (float)ksT_s[e * SQ + m];
      kcum_s[e] = s;
    }
    __syncthreads();

    // ---- out = (tril(S) @ v + qs @ ctx_prev) * dinv
    for (int i = 0; i < 4; ++i) {
      const int idx = wid * 4 + i, tm = idx >> 2, te = idx & 3;
      v8f c = v8f_zero();
      c = wmma_bf16(load_frag_a(S_s, SQ, tm * 16, 0, lane),
                    load_frag_b(v_s, SQ, 0, te * 16, lane), c);
      c = wmma_bf16(load_frag_a(S_s, SQ, tm * 16, 32, lane),
                    load_frag_b(v_s, SQ, 32, te * 16, lane), c);
      c = wmma_bf16(load_frag_a(qs_s, SQ, tm * 16, 0, lane),
                    load_frag_b(ctxb_s, SQ, 0, te * 16, lane), c);
      c = wmma_bf16(load_frag_a(qs_s, SQ, tm * 16, 32, lane),
                    load_frag_b(ctxb_s, SQ, 32, te * 16, lane), c);
#pragma unroll
      for (int r = 0; r < 8; ++r) {
        const int row = tm * 16 + r + ((lane >> 4) << 3);
        const int col = te * 16 + (lane & 15);
        out[(size_t)(b * 2048 + n0 + row) * 512 + h * 64 + col] = c[r] * dinv_s[row];
      }
    }
    __syncthreads();

    // ---- ctx += ks^T @ v
    for (int i = 0; i < 4; ++i) {
      const int idx = wid * 4 + i, td = idx >> 2, te = idx & 3;
      v8f c;
#pragma unroll
      for (int r = 0; r < 8; ++r) {
        const int d = td * 16 + r + ((lane >> 4) << 3);
        const int e = te * 16 + (lane & 15);
        c[r] = ctx_s[d * 68 + e];
      }
      c = wmma_bf16(load_frag_a(ksT_s, SQ, td * 16, 0, lane),
                    load_frag_b(v_s, SQ, 0, te * 16, lane), c);
      c = wmma_bf16(load_frag_a(ksT_s, SQ, td * 16, 32, lane),
                    load_frag_b(v_s, SQ, 32, te * 16, lane), c);
#pragma unroll
      for (int r = 0; r < 8; ++r) {
        const int d = td * 16 + r + ((lane >> 4) << 3);
        const int e = te * 16 + (lane & 15);
        ctx_s[d * 68 + e] = c[r];
        ctxb_s[d * SQ + e] = tobf(c[r]);
      }
    }
    __syncthreads();
  }
}

// ---------------------------------------------------------------------------
// out = LayerNorm(a + b) * g + beta ; 512 cols, 1 block/row; optional bf16 copy.
// ---------------------------------------------------------------------------
__global__ __launch_bounds__(256)
void add_layernorm_k(const float* __restrict__ a, const float* __restrict__ b,
                     const float* __restrict__ g, const float* __restrict__ be,
                     float* __restrict__ out, bf16_t* __restrict__ obf) {
  __shared__ float red[256];
  const int row = blockIdx.x, tid = threadIdx.x;
  const float* ap = a + (size_t)row * 512;
  const float* bp = b + (size_t)row * 512;
  const float x0 = ap[tid] + bp[tid];
  const float x1 = ap[tid + 256] + bp[tid + 256];
  red[tid] = x0 + x1;
  __syncthreads();
  for (int o = 128; o > 0; o >>= 1) {
    if (tid < o) red[tid] += red[tid + o];
    __syncthreads();
  }
  const float mu = red[0] * (1.0f / 512.0f);
  __syncthreads();
  const float d0 = x0 - mu, d1 = x1 - mu;
  red[tid] = d0 * d0 + d1 * d1;
  __syncthreads();
  for (int o = 128; o > 0; o >>= 1) {
    if (tid < o) red[tid] += red[tid + o];
    __syncthreads();
  }
  const float rs = rsqrtf(red[0] * (1.0f / 512.0f) + 1e-5f);
  const float y0 = d0 * rs * g[tid] + be[tid];
  const float y1 = d1 * rs * g[tid + 256] + be[tid + 256];
  out[(size_t)row * 512 + tid]       = y0;
  out[(size_t)row * 512 + tid + 256] = y1;
  if (obf) {
    obf[(size_t)row * 512 + tid]       = tobf(y0);
    obf[(size_t)row * 512 + tid + 256] = tobf(y1);
  }
}

// ---------------------------------------------------------------------------
// Column softmax over M for cross-attn keys: reads kv k-part fp32,
// writes transposed bf16 ksT[b][d][m] (B,512,2048).
// ---------------------------------------------------------------------------
__global__ __launch_bounds__(256)
void col_softmax_T_k(const float* __restrict__ kv, bf16_t* __restrict__ ksT) {
  __shared__ float red[256];
  const int blk = blockIdx.x, tid = threadIdx.x;
  const int b = blk >> 9, d = blk & 511;
  const float* base = kv + (size_t)b * 2048 * 1024 + d;
  float vals[8];
  float mx = -3.4e38f;
#pragma unroll
  for (int i = 0; i < 8; ++i) {
    vals[i] = base[(size_t)(tid + i * 256) * 1024];
    mx = fmaxf(mx, vals[i]);
  }
  red[tid] = mx;
  __syncthreads();
  for (int o = 128; o > 0; o >>= 1) {
    if (tid < o) red[tid] = fmaxf(red[tid], red[tid + o]);
    __syncthreads();
  }
  mx = red[0];
  __syncthreads();
  float s = 0.f;
#pragma unroll
  for (int i = 0; i < 8; ++i) {
    vals[i] = __expf(vals[i] - mx);
    s += vals[i];
  }
  red[tid] = s;
  __syncthreads();
  for (int o = 128; o > 0; o >>= 1) {
    if (tid < o) red[tid] += red[tid + o];
    __syncthreads();
  }
  const float inv = 1.0f / red[0];
  bf16_t* op = ksT + ((size_t)b * 512 + d) * 2048;
#pragma unroll
  for (int i = 0; i < 8; ++i) op[tid + i * 256] = tobf(vals[i] * inv);
}

// ---------------------------------------------------------------------------
// ctx[b][h] (64x64) = ks^T @ v over M=2048, async-staged bf16 WMMA,
// double-buffered 32-wide chunks.
// ---------------------------------------------------------------------------
__global__ __launch_bounds__(128)
void cross_ctx_k(const bf16_t* __restrict__ ksT, const bf16_t* __restrict__ kvbf,
                 float* __restrict__ ctx) {
  __shared__ __attribute__((aligned(16))) bf16_t a_s[2][64 * 40];  // [d][m]
  __shared__ __attribute__((aligned(16))) bf16_t b_s[2][32 * 72];  // [m][e]
  const int tid = threadIdx.x, lane = tid & 31, wid = tid >> 5;
  const int b = blockIdx.x >> 3, h = blockIdx.x & 7;
  const unsigned long long gA = (unsigned long long)ksT;
  const unsigned long long gB = (unsigned long long)kvbf;

  v8f acc[4];
#pragma unroll
  for (int i = 0; i < 4; ++i) acc[i] = v8f_zero();

  auto stage = [&](int buf, int m0) {  // 4 async ops / thread
#pragma unroll
    for (int i = 0; i < 2; ++i) {
      const int cid = i * 128 + tid;
      {  // a: 64 rows x 64B (4 chunks)
        const int r = cid >> 2, q = cid & 3;
        const unsigned lds = (unsigned)(size_t)(&a_s[buf][r * 40]) + q * 16;
        const unsigned gof =
            (unsigned)((((b * 512 + h * 64 + r) * 2048) + m0) * 2 + q * 16);
        async_b128(lds, gof, gA);
      }
      {  // b: 32 rows x 128B (8 chunks)
        const int r = cid >> 3, q = cid & 7;
        const unsigned lds = (unsigned)(size_t)(&b_s[buf][r * 72]) + q * 16;
        const unsigned gof =
            (unsigned)(((b * 2048 + m0 + r) * 1024 + 512 + h * 64) * 2 + q * 16);
        async_b128(lds, gof, gB);
      }
    }
  };

  stage(0, 0);
  for (int it = 0; it < 64; ++it) {
    const int cur = it & 1;
    if (it + 1 < 64) {
      stage(cur ^ 1, (it + 1) * 32);
      wait_async_le4();
    } else {
      wait_async_0();
    }
    __syncthreads();
    for (int i = 0; i < 4; ++i) {
      const int idx = wid * 4 + i, td = idx >> 2, te = idx & 3;
      acc[i] = wmma_bf16(load_frag_a(a_s[cur], 40, td * 16, 0, lane),
                         load_frag_b(b_s[cur], 72, 0, te * 16, lane), acc[i]);
    }
    __syncthreads();
  }
  for (int i = 0; i < 4; ++i) {
    const int idx = wid * 4 + i, td = idx >> 2, te = idx & 3;
#pragma unroll
    for (int r = 0; r < 8; ++r) {
      const int d = td * 16 + r + ((lane >> 4) << 3);
      const int e = te * 16 + (lane & 15);
      ctx[(((size_t)b * 8 + h) * 64 + d) * 64 + e] = acc[i][r];
    }
  }
}

// ---------------------------------------------------------------------------
// Cross-attn output: out = softmax(q)*E^-0.5 @ ctx, per (b,h,64-token chunk).
// ---------------------------------------------------------------------------
__global__ __launch_bounds__(128)
void cross_out_k(const float* __restrict__ qproj, const float* __restrict__ ctx,
                 float* __restrict__ out) {
  __shared__ __attribute__((aligned(16))) bf16_t qs_s[64 * SQ];
  __shared__ __attribute__((aligned(16))) bf16_t cb_s[64 * SQ];
  const int tid = threadIdx.x, lane = tid & 31, wid = tid >> 5;
  const int bi = blockIdx.x;
  const int b = bi >> 8, h = (bi >> 5) & 7, n0 = (bi & 31) * 64;

  for (int t = tid; t < 4096; t += 128) {
    const int d = t >> 6, e = t & 63;
    cb_s[d * SQ + e] = tobf(ctx[(((size_t)b * 8 + h) * 64 + d) * 64 + e]);
  }
  if (tid < 64) {
    const int m = tid;
    const size_t rb = (size_t)(b * 2048 + n0 + m) * 512 + h * 64;
    float mx = -3.4e38f;
    for (int e = 0; e < 64; ++e) mx = fmaxf(mx, qproj[rb + e]);
    float s = 0.f;
    for (int e = 0; e < 64; ++e) {
      const float ev = __expf(qproj[rb + e] - mx);
      s += ev;
      qs_s[m * SQ + e] = tobf(ev);
    }
    const float inv = 0.125f / s;
    for (int e = 0; e < 64; ++e)
      qs_s[m * SQ + e] = tobf((float)qs_s[m * SQ + e] * inv);
  }
  __syncthreads();

  for (int i = 0; i < 4; ++i) {
    const int idx = wid * 4 + i, tm = idx >> 2, te = idx & 3;
    v8f c = v8f_zero();
    c = wmma_bf16(load_frag_a(qs_s, SQ, tm * 16, 0, lane),
                  load_frag_b(cb_s, SQ, 0, te * 16, lane), c);
    c = wmma_bf16(load_frag_a(qs_s, SQ, tm * 16, 32, lane),
                  load_frag_b(cb_s, SQ, 32, te * 16, lane), c);
#pragma unroll
    for (int r = 0; r < 8; ++r) {
      const int row = tm * 16 + r + ((lane >> 4) << 3);
      const int col = te * 16 + (lane & 15);
      out[(size_t)(b * 2048 + n0 + row) * 512 + h * 64 + col] = c[r];
    }
  }
}

// ---------------------------------------------------------------------------
extern "C" void kernel_launch(void* const* d_in, const int* in_sizes, int n_in,
                              void* d_out, int out_size, void* d_ws, size_t ws_size,
                              hipStream_t stream) {
  (void)in_sizes; (void)n_in; (void)out_size; (void)ws_size;
  const float* x      = (const float*)d_in[0];
  const float* memory = (const float*)d_in[1];
  const float* W_qvk  = (const float*)d_in[2];
  const float* b_qvk  = (const float*)d_in[3];
  const float* W_kv   = (const float*)d_in[4];
  const float* b_kv   = (const float*)d_in[5];
  const float* W_q    = (const float*)d_in[6];
  const float* b_q    = (const float*)d_in[7];
  const float* W_ff1  = (const float*)d_in[8];
  const float* b_ff1  = (const float*)d_in[9];
  const float* W_ff2  = (const float*)d_in[10];
  const float* b_ff2  = (const float*)d_in[11];
  const float* ln1_g  = (const float*)d_in[12];
  const float* ln1_b  = (const float*)d_in[13];
  const float* ln2_g  = (const float*)d_in[14];
  const float* ln2_b  = (const float*)d_in[15];
  const float* ln3_g  = (const float*)d_in[16];
  const float* ln3_b  = (const float*)d_in[17];
  float* out = (float*)d_out;

  char* wsb = (char*)d_ws;
  size_t off = 0;
  auto alloc = [&](size_t bytes) {
    char* p = wsb + off;
    off += (bytes + 255) & ~(size_t)255;
    return p;
  };
  float*  qvk    = (float*)alloc((size_t)4096 * 1536 * 4);
  float*  attn1  = (float*)alloc((size_t)4096 * 512 * 4);
  float*  ln1    = (float*)alloc((size_t)4096 * 512 * 4);
  float*  kv     = (float*)alloc((size_t)4096 * 1024 * 4);
  float*  ctx    = (float*)alloc((size_t)2 * 8 * 64 * 64 * 4);
  float*  qproj  = (float*)alloc((size_t)4096 * 512 * 4);
  float*  attn2  = (float*)alloc((size_t)4096 * 512 * 4);
  float*  ln2    = (float*)alloc((size_t)4096 * 512 * 4);
  float*  ff2    = (float*)alloc((size_t)4096 * 512 * 4);
  bf16_t* x_bf   = (bf16_t*)alloc((size_t)4096 * 512 * 2);
  bf16_t* mem_bf = (bf16_t*)alloc((size_t)4096 * 512 * 2);
  bf16_t* Wqvk_bf = (bf16_t*)alloc((size_t)512 * 1536 * 2);
  bf16_t* Wkv_bf  = (bf16_t*)alloc((size_t)512 * 1024 * 2);
  bf16_t* Wq_bf   = (bf16_t*)alloc((size_t)512 * 512 * 2);
  bf16_t* Wff1_bf = (bf16_t*)alloc((size_t)512 * 2048 * 2);
  bf16_t* Wff2_bf = (bf16_t*)alloc((size_t)2048 * 512 * 2);
  bf16_t* ln1_bf  = (bf16_t*)alloc((size_t)4096 * 512 * 2);
  bf16_t* ln2_bf  = (bf16_t*)alloc((size_t)4096 * 512 * 2);
  bf16_t* ff1_bf  = (bf16_t*)alloc((size_t)4096 * 2048 * 2);
  bf16_t* kv_bf   = (bf16_t*)alloc((size_t)4096 * 1024 * 2);
  bf16_t* ksT_bf  = (bf16_t*)alloc((size_t)2 * 512 * 2048 * 2);

  // --- one-shot bf16 conversions (inputs + weights)
  cvt_f32_bf16_k<<<2048, 256, 0, stream>>>(x, x_bf);
  cvt_f32_bf16_k<<<2048, 256, 0, stream>>>(memory, mem_bf);
  cvt_f32_bf16_k<<<768, 256, 0, stream>>>(W_qvk, Wqvk_bf);
  cvt_f32_bf16_k<<<512, 256, 0, stream>>>(W_kv, Wkv_bf);
  cvt_f32_bf16_k<<<256, 256, 0, stream>>>(W_q, Wq_bf);
  cvt_f32_bf16_k<<<1024, 256, 0, stream>>>(W_ff1, Wff1_bf);
  cvt_f32_bf16_k<<<1024, 256, 0, stream>>>(W_ff2, Wff2_bf);

  // 1) qvk = x @ W_qvk + b_qvk
  gemm_bias_act<<<dim3(12, 32), 256, 0, stream>>>(x_bf, Wqvk_bf, b_qvk, qvk,
                                                  nullptr, 512, 1536, 0);
  // 2) causal linear self-attention (q,v,k column order per reference split)
  causal_linear_attn_k<<<16, 128, 0, stream>>>(qvk, attn1);
  // 3) ln1 = LN(attn1 + x)
  add_layernorm_k<<<4096, 256, 0, stream>>>(attn1, x, ln1_g, ln1_b, ln1, ln1_bf);
  // 4) kv = memory @ W_kv + b_kv  (fp32 for softmax, bf16 for WMMA consumers)
  gemm_bias_act<<<dim3(8, 32), 256, 0, stream>>>(mem_bf, Wkv_bf, b_kv, kv,
                                                 kv_bf, 512, 1024, 0);
  // 5) ks = softmax over M (per column), written transposed bf16
  col_softmax_T_k<<<1024, 256, 0, stream>>>(kv, ksT_bf);
  // 6) ctx = ks^T @ v per (b,h)
  cross_ctx_k<<<16, 128, 0, stream>>>(ksT_bf, kv_bf, ctx);
  // 7) qproj = ln1 @ W_q + b_q
  gemm_bias_act<<<dim3(4, 32), 256, 0, stream>>>(ln1_bf, Wq_bf, b_q, qproj,
                                                 nullptr, 512, 512, 0);
  // 8) attn2 = softmax(qproj)*E^-0.5 @ ctx
  cross_out_k<<<512, 128, 0, stream>>>(qproj, ctx, attn2);
  // 9) ln2 = LN(attn2 + ln1)
  add_layernorm_k<<<4096, 256, 0, stream>>>(attn2, ln1, ln2_g, ln2_b, ln2, ln2_bf);
  // 10) ff1 = relu(ln2 @ W_ff1 + b_ff1)  (bf16 only; fp32 copy not needed)
  gemm_bias_act<<<dim3(16, 32), 256, 0, stream>>>(ln2_bf, Wff1_bf, b_ff1,
                                                  nullptr, ff1_bf, 512, 2048, 1);
  // 11) ff2 = ff1 @ W_ff2 + b_ff2
  gemm_bias_act<<<dim3(4, 32), 256, 0, stream>>>(ff1_bf, Wff2_bf, b_ff2, ff2,
                                                 nullptr, 2048, 512, 0);
  // 12) out = LN(ff2 + ln2)
  add_layernorm_k<<<4096, 256, 0, stream>>>(ff2, ln2, ln3_g, ln3_b, out, nullptr);
}
